// HSoftmaxLayer_63316407877909
// MI455X (gfx1250) — compile-verified, matrix operand order checked by reading, not run on hardware
//
#include <hip/hip_runtime.h>
#include <hip/hip_bf16.h>

#define VOCAB 32000
#define DEPTH 18
#define INNER 31999   // VOCAB - 1
#define BT    64      // B*T = 4*16
#define DDIM  512
#define NT    4       // N-tiles per wave (A-operand reuse factor)

typedef __attribute__((ext_vector_type(16))) __bf16 v16bf;
typedef __attribute__((ext_vector_type(8)))  float  v8f;

// ---------------------------------------------------------------------------
// Kernel 1: hT[n*64 + bt] = sigmoid( dot(att[bt,:], W[n,:]) )
// One wave computes a 16(M=bt) x 64(N=inner) strip = 4 WMMA tiles, so the
// A-matrix (att) load + bf16 hi/lo split is amortized over 4 tiles.
// bf16-split WMMA: hi*hi + hi*lo + lo*hi in f32 accumulate => ~f32 accuracy.
// Strips = 4 (M) x 500 (N) = 2000 waves, 8 waves/block => 250 blocks.
// ---------------------------------------------------------------------------
__global__ __launch_bounds__(256)
void hsoftmax_gemm_kernel(const float* __restrict__ att,
                          const float* __restrict__ W,
                          float* __restrict__ hT) {
    __shared__ float lds[8][16][17];   // padded: stride 17 -> conflict-free

    const int lane   = threadIdx.x & 31;
    const int wave   = threadIdx.x >> 5;
    const int strip  = blockIdx.x * 8 + wave;
    const int mTile  = strip & 3;       // 4 M-tiles of 16 (BT=64)
    const int nGroup = strip >> 2;      // 500 groups of NT=4 N-tiles
    const int nl     = lane & 15;
    const int kb     = (lane < 16) ? 0 : 8;  // per-lane K base (16-bit A/B layout)

    const int mG = mTile * 16 + nl;           // att row this lane feeds
    const float* __restrict__ aRow = att + mG * DDIM;

    const float* __restrict__ bRow[NT];
#pragma unroll
    for (int t = 0; t < NT; ++t) {
        int nGr = nGroup * (16 * NT) + t * 16 + nl;   // W row this lane feeds
        bRow[t] = W + ((nGr < INNER) ? nGr : (INNER - 1)) * DDIM;
    }

    v8f acc[NT] = {};

    for (int kk = 0; kk < DDIM; kk += 32) {
        // ISA 16-bit A/B lane layout: e in [0,8) -> K = kb+e,
        //                             e in [8,16) -> K = kb+8+e (i.e. +16 block)
        v16bf ahi, alo;
#pragma unroll
        for (int e = 0; e < 16; ++e) {
            float av = aRow[kk + kb + ((e < 8) ? e : (e + 8))];
            __bf16 ah = (__bf16)av;
            ahi[e] = ah;
            alo[e] = (__bf16)(av - (float)ah);
        }

#pragma unroll
        for (int t = 0; t < NT; ++t) {
            v16bf bhi, blo;
#pragma unroll
            for (int e = 0; e < 16; ++e) {
                float bv = bRow[t][kk + kb + ((e < 8) ? e : (e + 8))];
                __bf16 bh = (__bf16)bv;
                bhi[e] = bh;
                blo[e] = (__bf16)(bv - (float)bh);
            }
            // D = A*B + C ; split-product accumulation for ~f32 precision
            acc[t] = __builtin_amdgcn_wmma_f32_16x16x32_bf16(
                         false, ahi, false, bhi, (short)0, acc[t], false, false);
            acc[t] = __builtin_amdgcn_wmma_f32_16x16x32_bf16(
                         false, ahi, false, blo, (short)0, acc[t], false, false);
            acc[t] = __builtin_amdgcn_wmma_f32_16x16x32_bf16(
                         false, alo, false, bhi, (short)0, acc[t], false, false);
        }
    }

    // Epilogue: sigmoid (fast v_rcp, no IEEE divide) + LDS transpose + store
    const int mBase = (lane < 16) ? 0 : 8;
#pragma unroll
    for (int t = 0; t < NT; ++t) {
        // C/D layout: lane L holds column n = L%16, rows m = r + (L<16 ? 0 : 8)
#pragma unroll
        for (int r = 0; r < 8; ++r) {
            float x = acc[t][r];
            float s = __builtin_amdgcn_rcpf(1.0f + __expf(-x));   // sigmoid
            lds[wave][nl][mBase + r] = s;
        }
        // Transposed, coalesced store: hT[n*64 + mTile*16 + m], 16 contiguous
        // dwords per half-wave (lanes 0-15 one n-row, lanes 16-31 the next).
#pragma unroll
        for (int p = 0; p < 8; ++p) {
            int nLoc = p * 2 + (lane >> 4);
            int mIdx = lane & 15;
            int nG   = nGroup * (16 * NT) + t * 16 + nLoc;
            if (nG < INNER)
                hT[nG * BT + mTile * 16 + mIdx] = lds[wave][nLoc][mIdx];
        }
    }
}

// ---------------------------------------------------------------------------
// Kernel 2: out[bt*VOCAB + v] = ln2 * sum_d log2(clip(hT[idx[v,d]]*sign+bias))
// One wave per vocab entry v; lanes = bt (two halves of 32). Each depth step:
// uniform idx/sign/bias reads + two coalesced 128B row loads from L2-resident
// hT, fma, clamp, one v_log_f32 per element.
// ---------------------------------------------------------------------------
__global__ __launch_bounds__(256)
void hsoftmax_path_kernel(const float* __restrict__ hT,
                          const int*   __restrict__ pidx,
                          const float* __restrict__ psign,
                          const float* __restrict__ pbias,
                          float* __restrict__ out) {
    const int lane = threadIdx.x & 31;
    const int wave = threadIdx.x >> 5;
    const int v    = blockIdx.x * 8 + wave;
    if (v >= VOCAB) return;

    const int base = v * DEPTH;
    float acc0 = 0.0f, acc1 = 0.0f;

#pragma unroll
    for (int d = 0; d < DEPTH; ++d) {
        int   id = pidx[base + d];        // uniform across wave -> one request
        float sg = psign[base + d];
        float bz = pbias[base + d];
        const float* __restrict__ row = hT + id * BT;
        float g0 = fmaf(row[lane],      sg, bz);
        float g1 = fmaf(row[lane + 32], sg, bz);
        // clip(g, 1e-9, 1-1e-9); 1-1e-9 == 1.0f in fp32 (matches f32 reference)
        g0 = fminf(fmaxf(g0, 1e-9f), 1.0f);
        g1 = fminf(fmaxf(g1, 1e-9f), 1.0f);
        acc0 += __log2f(g0);
        acc1 += __log2f(g1);
    }

    const float LN2 = 0.6931471805599453f;
    out[lane * VOCAB + v]        = acc0 * LN2;
    out[(lane + 32) * VOCAB + v] = acc1 * LN2;
}

extern "C" void kernel_launch(void* const* d_in, const int* in_sizes, int n_in,
                              void* d_out, int out_size, void* d_ws, size_t ws_size,
                              hipStream_t stream) {
    const float* att   = (const float*)d_in[0];   // [4,16,512]
    const float* W     = (const float*)d_in[1];   // [31999,512]
    const int*   pidx  = (const int*)  d_in[2];   // [576000]
    const float* psign = (const float*)d_in[3];   // [576000]
    const float* pbias = (const float*)d_in[4];   // [576000]
    float*       out   = (float*)d_out;           // [64,32000]
    float*       hT    = (float*)d_ws;            // [INNER][64] = 8.19 MB scratch

    // 2000 strips (4 M x 500 N-groups), 8 waves/block -> 250 blocks
    hsoftmax_gemm_kernel<<<250, 256, 0, stream>>>(att, W, hT);
    // 32000 vocab entries, 8 waves/block -> 4000 blocks
    hsoftmax_path_kernel<<<VOCAB / 8, 256, 0, stream>>>(hT, pidx, psign, pbias, out);
}
